// Part_Graph_77086073029172
// MI455X (gfx1250) — matrix operand order repaired and checked by reading, not compiled, not run on hardware
//
#include <hip/hip_runtime.h>
#include <hip/hip_bf16.h>
#include <math.h>

// ---------------------------------------------------------------------------
// CDNA5 (gfx1250) implementation. Dominant compute = 8x implicit-GEMM conv3x3
// (M=4608, K=2313, N=256) on v_wmma_f32_16x16x32_bf16, with the im2col A
// matrix pre-swizzled into WMMA operand layout (built once, shared by all 8
// maps), pure-load hot loop, and async-to-LDS prefetch of epilogue weights.
// ---------------------------------------------------------------------------

typedef __bf16 bf16x16 __attribute__((ext_vector_type(16)));
typedef float  f32x8   __attribute__((ext_vector_type(8)));

#define DEV __device__ __forceinline__

constexpr int N_  = 2;
constexpr int C_  = 256;
constexpr int H_  = 48;
constexpr int W_  = 48;
constexpr int HW  = H_ * W_;          // 2304
constexpr int HID = 10;
constexpr int KT   = 73;              // 72 xp K-tiles + 1 extra-channel tile
constexpr int NMAP = 8;               // 6 ctx maps + du + dl
constexpr int MT16 = (N_ * HW) / 16;  // 288 16-pixel A tiles
constexpr int MT32 = (N_ * HW) / 32;  // 144 32-pixel block tiles

// d_out layout (floats): p_new | du_att | dl_att | fdep_att
constexpr int DU_OFF = 7 * N_ * HID * HW;        // 322560
constexpr int DL_OFF = DU_OFF + N_ * 4 * HW;     // 340992
constexpr int FD_OFF = DL_OFF + N_ * 2 * HW;     // 350208

DEV unsigned short f2bf(float f) {
  union { float f; unsigned u; } v; v.f = f;
  unsigned u = v.u;
  unsigned r = u + 0x7FFFu + ((u >> 16) & 1u);   // round-to-nearest-even
  return (unsigned short)(r >> 16);
}
DEV float sigmoidf_(float x) { return 1.0f / (1.0f + __expf(-x)); }

DEV float coordv(int c8, int hw) {
  int h = hw / W_, w = hw % W_;
  const float s = 2.0f / 48.0f;
  switch (c8) {
    case 0: return w * s - 1.0f;            // xmin
    case 1: return h * s - 1.0f;            // ymin
    case 2: return (w + 1) * s - 1.0f;      // xmax
    case 3: return (h + 1) * s - 1.0f;      // ymax
    case 4: return (w + 0.5f) * s - 1.0f;   // xc
    case 5: return (h + 0.5f) * s - 1.0f;   // yc
    case 6: return 1.0f / 48.0f;
    default: return 1.0f / 48.0f;
  }
}

// CDNA5 async global->LDS copy (ASYNCcnt-tracked); lds_off = byte offset in LDS.
DEV void async_copy16(unsigned lds_off, const void* gptr) {
  asm volatile("global_load_async_to_lds_b128 %0, %1, off"
               :: "v"(lds_off), "v"(gptr) : "memory");
}
DEV void wait_async0() { asm volatile("s_wait_asynccnt 0x0" ::: "memory"); }

typedef __attribute__((address_space(3))) float lds_float_t;
DEV unsigned lds_offset_of(const float* p) {
  return (unsigned)(unsigned long long)(lds_float_t*)p;
}

// --------------------------------------------------------------------------
// Pack conv3x3 weights into bf16 WMMA B-operand layout.
// K ordering: K' = tap*256 + (cin-1) for xp channels (K' < 2304),
//             K' = 2304 + tap for the extra channel (cin = 0).
// B operand per lane: N = lane%16, K = (lane/16)*8 + jj + (jj>=8 ? 8 : 0)  (jj = 0..15).
// --------------------------------------------------------------------------
__global__ __launch_bounds__(512) void k_pack(const float* __restrict__ ctx_w1,
                                              const float* __restrict__ du_a1,
                                              const float* __restrict__ dl_a1,
                                              unsigned short* __restrict__ w1p) {
  int b   = blockIdx.x;                 // (j*73 + t)*16 + nt
  int j   = b / (KT * 16);
  int t   = (b / 16) % KT;
  int nt  = b & 15;
  int tid = threadIdx.x;                // 0..511
  int lane = tid >> 4, jj = tid & 15;
  int K = t * 32 + ((lane >> 4) << 3) + jj + ((jj >= 8) ? 8 : 0);
  int o = nt * 16 + (lane & 15);
  float val = 0.0f;
  int tap = -1, cin = 0;
  if (K < 2304)      { tap = K >> 8; cin = 1 + (K & 255); }
  else if (K < 2313) { tap = K - 2304; cin = 0; }
  if (tap >= 0) {
    if (j < 6)       val = ctx_w1[((size_t)(j * 256 + o) * 257 + cin) * 9 + tap];
    else if (j == 6) val = du_a1[((size_t)o * 257 + cin) * 9 + tap];
    else             val = dl_a1[((size_t)o * 257 + cin) * 9 + tap];
  }
  w1p[(size_t)b * 512 + tid] = f2bf(val);
}

// Transpose xp (N,C,HW) -> xpt (N,HW,C) for coalesced im2col gathers.
__global__ void k_xpt(const float* __restrict__ xp, float* __restrict__ xpt) {
  int idx = blockIdx.x * 256 + threadIdx.x;     // (n,c,hw)
  int hw = idx % HW; int q = idx / HW;
  int c = q % C_;    int n = q / C_;
  xpt[((size_t)n * HW + hw) * C_ + c] = xp[idx];
}

// --------------------------------------------------------------------------
// Build im2col A matrix, pre-swizzled into the WMMA A-operand lane layout.
// A operand: element (p,K) lives at lane = p + 16*((K>>3)&1),
//            jcol = (K&7) + (K>=16 ? 8 : 0). One 16x32 tile = 32 lanes x 16 bf16.
// Map-independent part (t = 0..71): shared by all 8 maps.
// --------------------------------------------------------------------------
__global__ __launch_bounds__(512) void k_im2col(const float* __restrict__ xpt,
                                                unsigned short* __restrict__ Abase) {
  int bid = blockIdx.x;                // mt16*72 + t
  int mt16 = bid / 72, t = bid % 72;
  int tid = threadIdx.x;               // lane*16 + jj
  int lane = tid >> 4, jj = tid & 15;
  int p = lane & 15, halfK = lane >> 4;
  int K = halfK * 8 + jj + ((jj >= 8) ? 8 : 0);
  int kp = t * 32 + K;                 // < 2304
  int tap = kp >> 8, cm1 = kp & 255;
  int m = mt16 * 16 + p;
  int n = m / HW, hw = m % HW;
  int h = hw / W_, w = hw % W_;
  int hh = h + tap / 3 - 1, ww = w + tap % 3 - 1;
  float val = 0.0f;
  if ((unsigned)hh < (unsigned)H_ && (unsigned)ww < (unsigned)W_)
    val = xpt[((size_t)n * HW + hh * W_ + ww) * C_ + cm1];
  Abase[(size_t)bid * 512 + tid] = f2bf(val);
}

// Map-dependent extra-channel A tile (t = 72): 9 taps of depatt / h_att.
__global__ __launch_bounds__(512) void k_im2col_ext(const float* __restrict__ depatt,
                                                    const float* __restrict__ h_att,
                                                    unsigned short* __restrict__ Aext) {
  int bid = blockIdx.x;                // j*288 + mt16
  int j = bid / MT16, mt16 = bid % MT16;
  int tid = threadIdx.x;
  int lane = tid >> 4, jj = tid & 15;
  int p = lane & 15, halfK = lane >> 4;
  int K = halfK * 8 + jj + ((jj >= 8) ? 8 : 0);
  int m = mt16 * 16 + p;
  int n = m / HW, hw = m % HW;
  float val = 0.0f;
  if (K < 9) {
    int h = hw / W_, w = hw % W_;
    int hh = h + K / 3 - 1, ww = w + K % 3 - 1;
    if ((unsigned)hh < (unsigned)H_ && (unsigned)ww < (unsigned)W_) {
      const float* extra = (j < 6) ? (depatt + (size_t)(j * N_ + n) * HW)
                                   : (h_att + (size_t)((j - 5) * N_ + n) * HW);
      val = extra[hh * W_ + ww];
    }
  }
  Aext[(size_t)bid * 512 + tid] = f2bf(val);
}

// key_f = relu(conv1x1(concat[xp, coord], dc_k_w, dc_k_b))   (N,64,HW)
__global__ void k_keyf(const float* __restrict__ xp, const float* __restrict__ kw,
                       const float* __restrict__ kb, float* __restrict__ keyf) {
  int idx = blockIdx.x * 256 + threadIdx.x;     // (n,d,hw)
  int hw = idx % HW; int q = idx / HW;
  int d = q % 64;    int n = q / 64;
  const float* wrow = kw + (size_t)d * 264;
  const float* src  = xp + (size_t)n * C_ * HW + hw;
  float acc = kb[d];
  for (int c = 0; c < C_; ++c) acc += wrow[c] * src[(size_t)c * HW];
  for (int c8 = 0; c8 < 8; ++c8) acc += wrow[256 + c8] * coordv(c8, hw);
  keyf[idx] = fmaxf(acc, 0.0f);
}

// px = dp_proj @ xp   (N,10,HW)
__global__ void k_px(const float* __restrict__ xp, const float* __restrict__ dp_proj,
                     float* __restrict__ px) {
  int idx = blockIdx.x * 256 + threadIdx.x;     // (n,o,hw)
  int hw = idx % HW; int q = idx / HW;
  int o = q % HID;   int n = q / HID;
  const float* wrow = dp_proj + (size_t)o * C_;
  const float* src  = xp + (size_t)n * C_ * HW + hw;
  float acc = 0.0f;
  for (int c = 0; c < C_; ++c) acc += wrow[c] * src[(size_t)c * HW];
  px[idx] = acc;
}

// Per (i,n): softmax(p_att[1+i]) -> attended context -> q -> scalar q.wq + bias
__global__ void k_ctx(const float* __restrict__ p_att, const float* __restrict__ xp,
                      const float* __restrict__ dc_q_w, const float* __restrict__ dc_q_b,
                      const float* __restrict__ dc_p_w, const float* __restrict__ dc_p_b,
                      float* __restrict__ qk) {
  int i = blockIdx.x >> 1, n = blockIdx.x & 1;
  int tid = threadIdx.x;
  __shared__ float sna[HW];
  __shared__ float sred[256];
  __shared__ float sqin[264];
  __shared__ float sq[64];
  const float* hu = p_att + (size_t)((1 + i) * N_ + n) * HW;
  float m = -3.4e38f;
  for (int k = tid; k < HW; k += 256) m = fmaxf(m, hu[k]);
  sred[tid] = m; __syncthreads();
  for (int s = 128; s > 0; s >>= 1) {
    if (tid < s) sred[tid] = fmaxf(sred[tid], sred[tid + s]);
    __syncthreads();
  }
  float mx = sred[0]; __syncthreads();
  float ssum = 0.0f;
  for (int k = tid; k < HW; k += 256) { float e = __expf(hu[k] - mx); sna[k] = e; ssum += e; }
  sred[tid] = ssum; __syncthreads();
  for (int s = 128; s > 0; s >>= 1) {
    if (tid < s) sred[tid] += sred[tid + s];
    __syncthreads();
  }
  float invZ = 1.0f / sred[0];
  __syncthreads();
  for (int c = tid; c < 264; c += 256) {
    float acc = 0.0f;
    if (c < 256) {
      const float* src = xp + (size_t)(n * C_ + c) * HW;
      for (int k = 0; k < HW; ++k) acc += src[k] * sna[k];
    } else {
      int c8 = c - 256;
      for (int k = 0; k < HW; ++k) acc += coordv(c8, k) * sna[k];
    }
    sqin[c] = acc * invZ;
  }
  __syncthreads();
  if (tid < 64) {
    float a = dc_q_b[tid];
    const float* wrow = dc_q_w + (size_t)tid * 264;
    for (int c = 0; c < 264; ++c) a += wrow[c] * sqin[c];
    sq[tid] = fmaxf(a, 0.0f);
  }
  __syncthreads();
  if (tid == 0) {
    float a = 0.0f;
    for (int d = 0; d < 64; ++d) a += sq[d] * dc_p_w[d];     // wq = dc_p_w[:64]
    qk[i * N_ + n] = a + dc_p_b[0];
  }
}

// dep_att[i,n,hw] = sigmoid(qk + wk.key_f) * (1 - p_att[1+i])
__global__ void k_depatt(const float* __restrict__ keyf, const float* __restrict__ qk,
                         const float* __restrict__ dc_p_w, const float* __restrict__ p_att,
                         float* __restrict__ depatt) {
  int idx = blockIdx.x * 256 + threadIdx.x;
  int hw = idx % HW; int g = idx / HW;     // g = i*2+n
  int n = g & 1, i = g >> 1;
  const float* kf = keyf + (size_t)(n * 64) * HW + hw;
  float e = qk[g];
  for (int d = 0; d < 64; ++d) e += dc_p_w[64 + d] * kf[(size_t)d * HW];
  float att = sigmoidf_(e);
  depatt[idx] = att * (1.0f - p_att[(size_t)((1 + i) * N_ + n) * HW + hw]);
}

// --------------------------------------------------------------------------
// THE WMMA kernel: conv3x3 (257ch -> 256ch), relu, fused conv1x1 -> S ch.
// Block = 32 pixels x all 256 out-channels. 8 waves x (2 M-tiles x 2 N-tiles).
// Hot loop: pure coalesced loads of pre-swizzled A/B + 4 WMMAs per K-step,
// all streams addressed as loop-affine offsets from invariant bases so LSR
// reduces them to pointer increments; prefetch is unconditional (speculative,
// over-run stays inside the workspace).
// Epilogue w2 fetched by async global->LDS DMA overlapped with the GEMM.
// --------------------------------------------------------------------------
__global__ __launch_bounds__(256) void k_conv(
    const unsigned short* __restrict__ Abase, const unsigned short* __restrict__ Aext,
    const unsigned short* __restrict__ w1p,
    const float* __restrict__ ctx_w2, const float* __restrict__ ctx_b2,
    const float* __restrict__ du_a2, const float* __restrict__ du_a2b,
    const float* __restrict__ dl_a2, const float* __restrict__ dl_a2b,
    float* __restrict__ out) {
  int mt = blockIdx.x, j = blockIdx.y;           // mt: 0..143, j: 0..7
  int tid = threadIdx.x, lane = tid & 31, wid = tid >> 5;
  __shared__ float sH[32 * 256];                  // relu(conv3x3) tile
  __shared__ __align__(16) float sW2[5 * 256];

  int S; const float* w2; const float* b2;
  if (j < 6)       { S = 5; w2 = ctx_w2 + (size_t)j * 5 * 256; b2 = ctx_b2 + j * 5; }
  else if (j == 6) { S = 4; w2 = du_a2; b2 = du_a2b; }
  else             { S = 2; w2 = dl_a2; b2 = dl_a2b; }

  // Kick off async DMA of w2 into LDS; completes while the GEMM runs.
  {
    unsigned base3 = lds_offset_of(sW2);
    int nch = S * 64;                             // 16B chunks
    for (int ch = tid; ch < nch; ch += 256)
      async_copy16(base3 + (unsigned)ch * 16, (const char*)w2 + (size_t)ch * 16);
  }

  const bf16x16* ap  = reinterpret_cast<const bf16x16*>(Abase);
  const bf16x16* aep = reinterpret_cast<const bf16x16*>(Aext);
  const bf16x16* bp  = reinterpret_cast<const bf16x16*>(w1p);

  size_t nt0 = (size_t)(wid * 2), nt1 = nt0 + 1;
  // Loop-invariant stream bases (per lane); offsets below are affine in t.
  const bf16x16* bb  = bp + ((size_t)j * KT * 16) * 32 + lane;      // + t*512 + nt*32
  const bf16x16* ab0 = ap + ((size_t)(mt * 2 + 0) * 72) * 32 + lane; // + t*32
  const bf16x16* ab1 = ap + ((size_t)(mt * 2 + 1) * 72) * 32 + lane;
  f32x8 acc00 = {}, acc01 = {}, acc10 = {}, acc11 = {};

#pragma unroll 4
  for (int t = 0; t < 72; ++t) {
    size_t toff = (size_t)t * 512;
    bf16x16 b0 = bb[toff + nt0 * 32];
    bf16x16 b1 = bb[toff + nt1 * 32];
    bf16x16 a0 = ab0[(size_t)t * 32];
    bf16x16 a1 = ab1[(size_t)t * 32];
    // Speculative prefetch ~8 K-tiles ahead on the B stream (branchless).
    __builtin_prefetch(&bb[toff + (size_t)8 * 512 + nt0 * 32], 0, 1);
    acc00 = __builtin_amdgcn_wmma_f32_16x16x32_bf16(false, a0, false, b0, (short)0, acc00, false, false);
    acc01 = __builtin_amdgcn_wmma_f32_16x16x32_bf16(false, a0, false, b1, (short)0, acc01, false, false);
    acc10 = __builtin_amdgcn_wmma_f32_16x16x32_bf16(false, a1, false, b0, (short)0, acc10, false, false);
    acc11 = __builtin_amdgcn_wmma_f32_16x16x32_bf16(false, a1, false, b1, (short)0, acc11, false, false);
  }
  { // t = 72: map-dependent extra-channel tile
    size_t toff = (size_t)72 * 512;
    bf16x16 b0 = bb[toff + nt0 * 32];
    bf16x16 b1 = bb[toff + nt1 * 32];
    bf16x16 a0 = aep[((size_t)j * MT16 + (mt * 2 + 0)) * 32 + lane];
    bf16x16 a1 = aep[((size_t)j * MT16 + (mt * 2 + 1)) * 32 + lane];
    acc00 = __builtin_amdgcn_wmma_f32_16x16x32_bf16(false, a0, false, b0, (short)0, acc00, false, false);
    acc01 = __builtin_amdgcn_wmma_f32_16x16x32_bf16(false, a0, false, b1, (short)0, acc01, false, false);
    acc10 = __builtin_amdgcn_wmma_f32_16x16x32_bf16(false, a1, false, b0, (short)0, acc10, false, false);
    acc11 = __builtin_amdgcn_wmma_f32_16x16x32_bf16(false, a1, false, b1, (short)0, acc11, false, false);
  }

  // C/D layout: VGPR r, lane l -> pixel (r + 8*(l/16)), out-chan (tile*16 + l%16)
  int prow = (lane >> 4) * 8;
  int ncol = lane & 15;
  int nc0 = (int)nt0 * 16 + ncol, nc1 = (int)nt1 * 16 + ncol;
#pragma unroll
  for (int r = 0; r < 8; ++r) {
    sH[(prow + r) * 256 + nc0]      = fmaxf(acc00[r], 0.0f);
    sH[(prow + r) * 256 + nc1]      = fmaxf(acc01[r], 0.0f);
    sH[(16 + prow + r) * 256 + nc0] = fmaxf(acc10[r], 0.0f);
    sH[(16 + prow + r) * 256 + nc1] = fmaxf(acc11[r], 0.0f);
  }
  wait_async0();        // w2 DMA done (waves with no issued copies pass trivially)
  __syncthreads();      // publish sH + sW2
  int mbase = mt * 32;
  for (int idx = tid; idx < 32 * S; idx += 256) {
    int p = idx / S, s = idx - p * S;
    float acc = b2[s];
    const float* wrow = &sW2[s * 256];
    const float* hrow = &sH[p * 256];
    for (int o = 0; o < 256; ++o) acc += wrow[o] * hrow[o];
    int m = mbase + p;
    int nn = m / HW, hw = m % HW;
    if (j < 6)       out[FD_OFF + (size_t)((j * N_ + nn) * 5 + s) * HW + hw] = acc;
    else if (j == 6) out[DU_OFF + (size_t)(nn * 4 + s) * HW + hw] = acc;
    else             out[DL_OFF + (size_t)(nn * 2 + s) * HW + hw] = acc;
  }
}

// Per-pixel softmaxes: att_sm (5ch), da_u (4ch), da_l (2ch)
__global__ void k_smax(const float* __restrict__ out, float* __restrict__ attsm,
                       float* __restrict__ dau, float* __restrict__ dal) {
  int idx = blockIdx.x * 256 + threadIdx.x;
  const int T1 = 12 * HW, T2 = N_ * HW;
  if (idx < T1) {
    int g = idx / HW, hw = idx % HW;
    const float* src = out + FD_OFF + (size_t)g * 5 * HW + hw;
    float v[5], mx = -3.4e38f;
    for (int s = 0; s < 5; ++s) { v[s] = src[(size_t)s * HW]; mx = fmaxf(mx, v[s]); }
    float Z = 0.0f;
    for (int s = 0; s < 5; ++s) { v[s] = __expf(v[s] - mx); Z += v[s]; }
    float iZ = 1.0f / Z;
    float* dst = attsm + (size_t)g * 5 * HW + hw;
    for (int s = 0; s < 5; ++s) dst[(size_t)s * HW] = v[s] * iZ;
  } else if (idx < T1 + T2) {
    int q = idx - T1; int hw = q % HW; int nn = q / HW;
    const float* src = out + DU_OFF + (size_t)nn * 4 * HW + hw;
    float v[4], mx = -3.4e38f;
    for (int s = 0; s < 4; ++s) { v[s] = src[(size_t)s * HW]; mx = fmaxf(mx, v[s]); }
    float Z = 0.0f;
    for (int s = 0; s < 4; ++s) { v[s] = __expf(v[s] - mx); Z += v[s]; }
    float iZ = 1.0f / Z;
    float* dst = dau + (size_t)nn * 4 * HW + hw;
    for (int s = 0; s < 4; ++s) dst[(size_t)s * HW] = v[s] * iZ;
  } else if (idx < T1 + 2 * T2) {
    int q = idx - T1 - T2; int hw = q % HW; int nn = q / HW;
    const float* src = out + DL_OFF + (size_t)nn * 2 * HW + hw;
    float v0 = src[0], v1 = src[HW];
    float mx = fmaxf(v0, v1);
    float e0 = __expf(v0 - mx), e1 = __expf(v1 - mx);
    float iZ = 1.0f / (e0 + e1);
    float* dst = dal + (size_t)nn * 2 * HW + hw;
    dst[0] = e0 * iZ; dst[HW] = e1 * iZ;
  }
}

// decomp "rel": parts = relu(conv1x1(relu(conv3x3(concat[parent*att_i*pat, child]))))
__global__ void k_rel(const float* __restrict__ h_nodes, const float* __restrict__ h_att,
                      const float* __restrict__ p_nodes,
                      const float* __restrict__ dau, const float* __restrict__ dal,
                      const float* __restrict__ du_r1, const float* __restrict__ du_r2,
                      const float* __restrict__ dl_r1, const float* __restrict__ dl_r2,
                      float* __restrict__ parts) {
  __shared__ float s_r1[2][1800];
  __shared__ float s_r2[2][100];
  int tid = threadIdx.x;
  for (int i2 = tid; i2 < 1800; i2 += 256) { s_r1[0][i2] = du_r1[i2]; s_r1[1][i2] = dl_r1[i2]; }
  for (int i2 = tid; i2 < 100; i2 += 256)  { s_r2[0][i2] = du_r2[i2]; s_r2[1][i2] = dl_r2[i2]; }
  __syncthreads();
  int idx = blockIdx.x * 256 + tid;          // (ch, n, hw), ch 0..5
  int hw = idx % HW; int q = idx / HW; int n = q & 1; int ch = q >> 1;
  int grp = (ch < 4) ? 0 : 1;
  const float* parent = h_nodes + (size_t)((grp + 1) * N_ + n) * HID * HW;
  const float* pat    = h_att + (size_t)((grp + 1) * N_ + n) * HW;
  const float* child  = p_nodes + (size_t)((1 + ch) * N_ + n) * HID * HW;
  const float* attm   = (grp == 0) ? (dau + (size_t)(n * 4 + ch) * HW)
                                   : (dal + (size_t)(n * 2 + (ch - 4)) * HW);
  int h = hw / W_, w = hw % W_;
  float y[10];
#pragma unroll
  for (int o = 0; o < 10; ++o) y[o] = 0.0f;
  for (int tap = 0; tap < 9; ++tap) {
    int hh = h + tap / 3 - 1, ww = w + tap % 3 - 1;
    if ((unsigned)hh < (unsigned)H_ && (unsigned)ww < (unsigned)W_) {
      int hw2 = hh * W_ + ww;
      float sc = attm[hw2] * pat[hw2];
      for (int cin = 0; cin < 20; ++cin) {
        float v = (cin < 10) ? parent[(size_t)cin * HW + hw2] * sc
                             : child[(size_t)(cin - 10) * HW + hw2];
#pragma unroll
        for (int o = 0; o < 10; ++o) y[o] += s_r1[grp][(o * 20 + cin) * 9 + tap] * v;
      }
    }
  }
  float yr[10];
#pragma unroll
  for (int c = 0; c < 10; ++c) yr[c] = fmaxf(y[c], 0.0f);
#pragma unroll
  for (int o = 0; o < 10; ++o) {
    float acc = 0.0f;
    for (int c = 0; c < 10; ++c) acc += s_r2[grp][o * 10 + c] * yr[c];
    parts[(size_t)(q * HID + o) * HW + hw] = fmaxf(acc, 0.0f);
  }
}

// Edge messages + deterministic segment-sum: xpp[b] = sum over 5 edges into b.
__global__ void k_msg(const float* __restrict__ attsm, const float* __restrict__ depatt,
                      const float* __restrict__ px, const float* __restrict__ p_nodes,
                      const float* __restrict__ dp_r1, const float* __restrict__ dp_r2,
                      const float* __restrict__ dp_r2b, float* __restrict__ xpp) {
  __shared__ float s1[1800], s2[100], sb[10];
  int tid = threadIdx.x;
  for (int i2 = tid; i2 < 1800; i2 += 256) s1[i2] = dp_r1[i2];
  for (int i2 = tid; i2 < 100; i2 += 256)  s2[i2] = dp_r2[i2];
  if (tid < 10) sb[tid] = dp_r2b[tid];
  __syncthreads();
  int idx = blockIdx.x * 256 + tid;          // (b, n, hw), b 0..5
  int hw = idx % HW; int q = idx / HW; int n = q & 1; int b = q >> 1;
  const float* hv  = p_nodes + (size_t)(b * N_ + n) * HID * HW;
  const float* pxn = px + (size_t)n * HID * HW;
  int h = hw / W_, w = hw % W_;
  float xacc[10];
#pragma unroll
  for (int o = 0; o < 10; ++o) xacc[o] = 0.0f;
  for (int a = 0; a < 6; ++a) {
    if (a == b) continue;
    int ce = (b < a) ? b : b - 1;
    const float* am = attsm + (size_t)((a * N_ + n) * 5 + ce) * HW;
    const float* dm = depatt + (size_t)(a * N_ + n) * HW;
    float y[10];
#pragma unroll
    for (int o = 0; o < 10; ++o) y[o] = 0.0f;
    for (int tap = 0; tap < 9; ++tap) {
      int hh = h + tap / 3 - 1, ww = w + tap % 3 - 1;
      if ((unsigned)hh < (unsigned)H_ && (unsigned)ww < (unsigned)W_) {
        int hw2 = hh * W_ + ww;
        float sc = am[hw2] * dm[hw2];
        for (int cin = 0; cin < 20; ++cin) {
          float v = (cin < 10) ? fmaxf(sc * pxn[(size_t)cin * HW + hw2], 0.0f)
                               : hv[(size_t)(cin - 10) * HW + hw2];
#pragma unroll
          for (int o = 0; o < 10; ++o) y[o] += s1[(o * 20 + cin) * 9 + tap] * v;
        }
      }
    }
#pragma unroll
    for (int o = 0; o < 10; ++o) {
      float acc = sb[o];
      for (int c = 0; c < 10; ++c) acc += s2[o * 10 + c] * fmaxf(y[c], 0.0f);
      xacc[o] += sigmoidf_(acc);
    }
  }
#pragma unroll
  for (int o = 0; o < 10; ++o) xpp[(size_t)(q * HID + o) * HW + hw] = xacc[o];
}

// GRU: p_new[t] from x_stack[t] and p_nodes[t]
__global__ void k_gru(const float* __restrict__ h_nodes, const float* __restrict__ p_nodes,
                      const float* __restrict__ parts, const float* __restrict__ xpp,
                      const float* __restrict__ gru_gw, const float* __restrict__ gru_gb,
                      const float* __restrict__ gru_cw, const float* __restrict__ alpha,
                      float* __restrict__ out) {
  __shared__ float sgw[280], sgb[14], scw[1400];
  int tid = threadIdx.x;
  for (int i2 = tid; i2 < 280; i2 += 256)  sgw[i2] = gru_gw[i2];
  for (int i2 = tid; i2 < 1400; i2 += 256) scw[i2] = gru_cw[i2];
  if (tid < 14) sgb[tid] = gru_gb[tid];
  __syncthreads();
  int idx = blockIdx.x * 256 + tid;          // (t, n, hw), t 0..6
  int hw = idx % HW; int q = idx / HW; int n = q & 1; int t = q >> 1;
  float al = alpha[0];
  float x[10], hv[10];
  const float* hp = p_nodes + (size_t)q * HID * HW;
#pragma unroll
  for (int c = 0; c < 10; ++c) hv[c] = hp[(size_t)c * HW + hw];
  if (t == 0) {
    const float* xb = h_nodes + (size_t)n * HID * HW;   // h_nodes[0]
#pragma unroll
    for (int c = 0; c < 10; ++c) x[c] = xb[(size_t)c * HW + hw];
  } else {
    const float* pb = parts + (size_t)((t - 1) * N_ + n) * HID * HW;
    const float* xb = xpp + (size_t)((t - 1) * N_ + n) * HID * HW;
#pragma unroll
    for (int c = 0; c < 10; ++c)
      x[c] = pb[(size_t)c * HW + hw] + al * xb[(size_t)c * HW + hw];
  }
  float g0 = sgb[t * 2 + 0], g1 = sgb[t * 2 + 1];
#pragma unroll
  for (int c = 0; c < 10; ++c) {
    g0 += sgw[(t * 2 + 0) * 20 + c] * x[c] + sgw[(t * 2 + 0) * 20 + 10 + c] * hv[c];
    g1 += sgw[(t * 2 + 1) * 20 + c] * x[c] + sgw[(t * 2 + 1) * 20 + 10 + c] * hv[c];
  }
  float r = sigmoidf_(g0), u = sigmoidf_(g1);
#pragma unroll
  for (int o = 0; o < 10; ++o) {
    float a2 = 0.0f;
    for (int c = 0; c < 10; ++c)
      a2 += scw[(t * 10 + o) * 20 + c] * x[c] + scw[(t * 10 + o) * 20 + 10 + c] * (r * hv[c]);
    float cnm = fmaxf(a2, 0.0f);
    out[(size_t)(q * HID + o) * HW + hw] = (1.0f - u) * hv[o] + u * cnm;
  }
}

// --------------------------------------------------------------------------
extern "C" void kernel_launch(void* const* d_in, const int* in_sizes, int n_in,
                              void* d_out, int out_size, void* d_ws, size_t ws_size,
                              hipStream_t stream) {
  (void)in_sizes; (void)n_in; (void)out_size; (void)ws_size;
  // d_in[0] = f_nodes (unused by the reference computation)
  const float* h_nodes = (const float*)d_in[1];
  const float* p_nodes = (const float*)d_in[2];
  const float* xp      = (const float*)d_in[3];
  const float* h_att   = (const float*)d_in[4];
  const float* p_att   = (const float*)d_in[5];
  const float* dc_q_w  = (const float*)d_in[6];
  const float* dc_q_b  = (const float*)d_in[7];
  const float* dc_k_w  = (const float*)d_in[8];
  const float* dc_k_b  = (const float*)d_in[9];
  const float* dc_p_w  = (const float*)d_in[10];
  const float* dc_p_b  = (const float*)d_in[11];
  const float* ctx_w1  = (const float*)d_in[12];
  const float* ctx_w2  = (const float*)d_in[13];
  const float* ctx_b2  = (const float*)d_in[14];
  const float* du_a1   = (const float*)d_in[15];
  const float* du_a2   = (const float*)d_in[16];
  const float* du_a2b  = (const float*)d_in[17];
  const float* du_r1   = (const float*)d_in[18];
  const float* du_r2   = (const float*)d_in[19];
  const float* dl_a1   = (const float*)d_in[20];
  const float* dl_a2   = (const float*)d_in[21];
  const float* dl_a2b  = (const float*)d_in[22];
  const float* dl_r1   = (const float*)d_in[23];
  const float* dl_r2   = (const float*)d_in[24];
  const float* dp_proj = (const float*)d_in[25];
  const float* dp_r1   = (const float*)d_in[26];
  const float* dp_r2   = (const float*)d_in[27];
  const float* dp_r2b  = (const float*)d_in[28];
  const float* gru_gw  = (const float*)d_in[29];
  const float* gru_gb  = (const float*)d_in[30];
  const float* gru_cw  = (const float*)d_in[31];
  const float* alpha   = (const float*)d_in[32];
  float* out = (float*)d_out;

  char* ws = (char*)d_ws;
  size_t off = 0;
  auto carve = [&](size_t bytes) -> char* {
    char* p = ws + off;
    off = (off + bytes + 255) & ~(size_t)255;
    return p;
  };
  unsigned short* w1p   = (unsigned short*)carve((size_t)NMAP * KT * 16 * 512 * 2);
  unsigned short* Abase = (unsigned short*)carve((size_t)MT16 * 72 * 512 * 2);
  unsigned short* Aext  = (unsigned short*)carve((size_t)NMAP * MT16 * 512 * 2);
  float* xpt    = (float*)carve((size_t)N_ * HW * C_ * 4);
  float* keyf   = (float*)carve((size_t)N_ * 64 * HW * 4);
  float* px     = (float*)carve((size_t)N_ * HID * HW * 4);
  float* qk     = (float*)carve(64);
  float* depatt = (float*)carve((size_t)6 * N_ * HW * 4);
  float* attsm  = (float*)carve((size_t)12 * 5 * HW * 4);
  float* dau    = (float*)carve((size_t)N_ * 4 * HW * 4);
  float* dal    = (float*)carve((size_t)N_ * 2 * HW * 4);
  float* parts  = (float*)carve((size_t)6 * N_ * HID * HW * 4);
  float* xpp    = (float*)carve((size_t)6 * N_ * HID * HW * 4);

  k_pack<<<NMAP * KT * 16, 512, 0, stream>>>(ctx_w1, du_a1, dl_a1, w1p);
  k_xpt<<<(N_ * HW * C_) / 256, 256, 0, stream>>>(xp, xpt);
  k_keyf<<<(N_ * 64 * HW) / 256, 256, 0, stream>>>(xp, dc_k_w, dc_k_b, keyf);
  k_px<<<(N_ * HID * HW) / 256, 256, 0, stream>>>(xp, dp_proj, px);
  k_ctx<<<12, 256, 0, stream>>>(p_att, xp, dc_q_w, dc_q_b, dc_p_w, dc_p_b, qk);
  k_depatt<<<(6 * N_ * HW) / 256, 256, 0, stream>>>(keyf, qk, dc_p_w, p_att, depatt);
  k_im2col<<<MT16 * 72, 512, 0, stream>>>(xpt, Abase);
  k_im2col_ext<<<NMAP * MT16, 512, 0, stream>>>(depatt, h_att, Aext);
  dim3 gconv(MT32, NMAP);
  k_conv<<<gconv, 256, 0, stream>>>(Abase, Aext, w1p, ctx_w2, ctx_b2,
                                    du_a2, du_a2b, dl_a2, dl_a2b, out);
  k_smax<<<(12 * HW + 2 * N_ * HW) / 256, 256, 0, stream>>>(out, attsm, dau, dal);
  k_rel<<<(6 * N_ * HW) / 256, 256, 0, stream>>>(h_nodes, h_att, p_nodes, dau, dal,
                                                 du_r1, du_r2, dl_r1, dl_r2, parts);
  k_msg<<<(6 * N_ * HW) / 256, 256, 0, stream>>>(attsm, depatt, px, p_nodes,
                                                 dp_r1, dp_r2, dp_r2b, xpp);
  k_gru<<<(7 * N_ * HW) / 256, 256, 0, stream>>>(h_nodes, p_nodes, parts, xpp,
                                                 gru_gw, gru_gb, gru_cw, alpha, out);
}
